// AdaptiveSTSamplingMixing_7791070675494
// MI455X (gfx1250) — compile-verified, matrix operand order checked by reading, not run on hardware
//
#include <hip/hip_runtime.h>
#include <cstdint>

typedef __attribute__((ext_vector_type(16))) _Float16 v16h;
typedef __attribute__((ext_vector_type(8)))  float    v8f;
typedef __attribute__((ext_vector_type(4)))  float    f32x4;
typedef int v4i __attribute__((ext_vector_type(4)));

// pointer-to-AS1-int4 and pointer-to-AS3-int4 (what the async builtin wants)
typedef __attribute__((address_space(1))) v4i* gv4ip;
typedef __attribute__((address_space(3))) v4i* lv4ip;

#if __has_builtin(__builtin_amdgcn_global_load_async_to_lds_b128) && \
    __has_builtin(__builtin_amdgcn_s_wait_asynccnt)
#define HAVE_ASYNC 1
#else
#define HAVE_ASYNC 0
#endif

#define GEMM_WAVES 4   // waves per GEMM block

// ---------------------------------------------------------------------------
// Generic skinny-M GEMM:  C[chunk] = A(MxK) @ B(KxN) (+bias if kc==1)
// f32 in/out, f16 WMMA core (v_wmma_f32_16x16x32_f16), f32 accumulation.
// One wave computes a 16x64 tile. B tiles (32x64 f32 = 8KB) are streamed with
// global_load_async_to_lds_b128 into a per-wave double buffer and consumed
// from LDS via ds_load (bank-conflict-free: bank index == column).
// split-K via kc chunks; each chunk writes its own M*N slice (deterministic).
// Requires: M%16==0, N%64==0, (K/kc)%32==0.
// ---------------------------------------------------------------------------
__global__ __launch_bounds__(32 * GEMM_WAVES) void gemm_wmma_kernel(
    const float* __restrict__ A, const float* __restrict__ Bm,
    const float* __restrict__ bias, float* __restrict__ C,
    int M, int N, int K, int kc)
{
#if HAVE_ASYNC
  __shared__ float lbuf[GEMM_WAVES][2][32 * 64];   // 64 KB
#endif
  const int lane = threadIdx.x & 31;
  const int wid  = blockIdx.x * (blockDim.x >> 5) + (threadIdx.x >> 5);
  const int nt64 = N >> 6;
  const int tilesTotal = (M >> 4) * nt64;
  const int chunk = wid / tilesTotal;
  if (chunk >= kc) return;
  const int tile = wid - chunk * tilesTotal;
  const int mt = tile / nt64;
  const int nt = tile - mt * nt64;

  const int hf  = lane >> 4;   // 0: lanes 0-15, 1: lanes 16-31
  const int idx = lane & 15;

  const long row  = (long)mt * 16 + idx;     // A row handled by this lane
  const long col0 = (long)nt * 64;

  const int Kc   = K / kc;
  const int kbeg = chunk * Kc;
  const int kend = kbeg + Kc;

  v8f acc[4] = {};
  const float* Arow = A + row * (long)K;

#if HAVE_ASYNC
  const int wslot = threadIdx.x >> 5;
  const int lrow  = lane >> 4;          // sub-row within a 512B async issue
  const int lc4   = (lane & 15) * 4;    // 4-float column offset
  auto issueB = [&](int kk, int bsel) {
    const float* gbase = Bm + (long)(kk + lrow) * N + col0 + lc4;
    #pragma unroll
    for (int i = 0; i < 16; ++i) {   // 16 x 512B = 32x64 f32 tile
      __builtin_amdgcn_global_load_async_to_lds_b128(
          (gv4ip)(uintptr_t)(gbase + (long)(2 * i) * N),
          (lv4ip)(unsigned)(uintptr_t)&lbuf[wslot][bsel][(lrow + 2 * i) * 64 + lc4],
          0, 0);
    }
  };
  issueB(kbeg, 0);
  if (kbeg + 32 < kend) issueB(kbeg + 32, 1);
  int cur = 0;
#endif

  for (int k0 = kbeg; k0 < kend; k0 += 32) {
    // ---- A tile: 16x32 f16.  lane(idx,hf), element e -> k = e + 8*hf + (e>=8 ? 8 : 0)
    const float* ap = Arow + k0 + hf * 8;
    f32x4 a0 = *(const f32x4*)(ap);
    f32x4 a1 = *(const f32x4*)(ap + 4);
    f32x4 a2 = *(const f32x4*)(ap + 16);
    f32x4 a3 = *(const f32x4*)(ap + 20);
    if (k0 + 32 < kend) __builtin_prefetch(Arow + k0 + 32, 0, 0);
    v16h av = {};
    av[0]  = (_Float16)a0[0]; av[1]  = (_Float16)a0[1];
    av[2]  = (_Float16)a0[2]; av[3]  = (_Float16)a0[3];
    av[4]  = (_Float16)a1[0]; av[5]  = (_Float16)a1[1];
    av[6]  = (_Float16)a1[2]; av[7]  = (_Float16)a1[3];
    av[8]  = (_Float16)a2[0]; av[9]  = (_Float16)a2[1];
    av[10] = (_Float16)a2[2]; av[11] = (_Float16)a2[3];
    av[12] = (_Float16)a3[0]; av[13] = (_Float16)a3[1];
    av[14] = (_Float16)a3[2]; av[15] = (_Float16)a3[3];

#if HAVE_ASYNC
    // wait until the current buffer's 16 async loads have landed in LDS
    if (k0 + 32 < kend) __builtin_amdgcn_s_wait_asynccnt(16);
    else                __builtin_amdgcn_s_wait_asynccnt(0);
    #pragma unroll
    for (int s = 0; s < 4; ++s) {
      const int n = s * 16 + idx;
      v16h bv = {};
      #pragma unroll
      for (int e = 0; e < 8; ++e)
        bv[e]     = (_Float16)lbuf[wslot][cur][(e + 8 * hf) * 64 + n];
      #pragma unroll
      for (int e = 0; e < 8; ++e)
        bv[8 + e] = (_Float16)lbuf[wslot][cur][(16 + e + 8 * hf) * 64 + n];
      acc[s] = __builtin_amdgcn_wmma_f32_16x16x32_f16(
          false, av, false, bv, (short)0, acc[s], false, false);
    }
    if (k0 + 64 < kend) issueB(k0 + 64, cur);   // refill consumed buffer
    cur ^= 1;
#else
    const float* bb = Bm + (long)(k0 + hf * 8) * N + col0 + idx;
    if (k0 + 32 < kend)
      __builtin_prefetch(Bm + (long)(k0 + 32 + hf * 8) * N + col0 + idx, 0, 0);
    #pragma unroll
    for (int s = 0; s < 4; ++s) {
      const float* bp  = bb + s * 16;
      const float* bp2 = bp + (long)16 * N;
      v16h bv = {};
      #pragma unroll
      for (int e = 0; e < 8; ++e) bv[e]     = (_Float16)bp[(long)e * N];
      #pragma unroll
      for (int e = 0; e < 8; ++e) bv[8 + e] = (_Float16)bp2[(long)e * N];
      acc[s] = __builtin_amdgcn_wmma_f32_16x16x32_f16(
          false, av, false, bv, (short)0, acc[s], false, false);
    }
#endif
  }

  // ---- store: lane(idx,hf): col = col0+s*16+idx, row(r) = mt*16 + r + 8*hf
  float* Cw = C + (long)chunk * M * N;
  #pragma unroll
  for (int s = 0; s < 4; ++s) {
    const long col = col0 + s * 16 + idx;
    const float bval = (bias != nullptr) ? bias[col] : 0.0f;
    #pragma unroll
    for (int r = 0; r < 8; ++r) {
      const long rr = (long)mt * 16 + r + hf * 8;
      Cw[rr * N + col] = acc[s][r] + bval;
    }
  }
}

// ---------------------------------------------------------------------------
// Bilinear multi-level sampling. One block per (b,q,g); thread = (sp, t).
// Produces spatial_feats (blk,32,64) = mean over T, temporal_feats (blk,8,64)
// = mean over SP. Deterministic LDS chunked reduction (no atomics).
// ---------------------------------------------------------------------------
__global__ __launch_bounds__(256) void sample_kernel(
    const float* __restrict__ f0, const float* __restrict__ f1,
    const float* __restrict__ f2, const float* __restrict__ f3,
    const float* __restrict__ boxes, const float* __restrict__ off,
    float* __restrict__ sf, float* __restrict__ tfeat)
{
  __shared__ float sbuf[256 * 8];            // (sp*8+t) x 8 channels
  const int blk = blockIdx.x;                // (b*100+q)*4 + g
  const int b   = blk / 400;
  const int rem = blk - b * 400;
  const int q   = rem >> 2;
  const int g   = rem & 3;
  const int tid = threadIdx.x;
  const int sp  = tid >> 3;
  const int t   = tid & 7;

  const float* box = boxes + ((long)b * 100 + q) * 4;
  const float cx = box[0], cy = box[1], zb = box[2], rr = box[3];
  const float sc  = exp2f(zb);
  const float whx = sc * exp2f(-0.5f * rr);
  const float why = sc * exp2f( 0.5f * rr);

  const float* op = off + (((long)b * 100 + q) * 128 + g * 32 + sp) * 3;
  const float x = cx + op[0] * whx;
  const float y = cy + op[1] * why;
  const float mapped = (zb + op[2]) - 2.0f;  // log2(stride0=4) == 2

  // softmax level weights: softmax(-(mapped - l)^2 / TAU), TAU = 2
  float lw[4];
  {
    float e[4], mx = -1e30f;
    #pragma unroll
    for (int l = 0; l < 4; ++l) { float d = mapped - (float)l; e[l] = -0.5f * d * d; mx = fmaxf(mx, e[l]); }
    float s = 0.0f;
    #pragma unroll
    for (int l = 0; l < 4; ++l) { e[l] = __expf(e[l] - mx); s += e[l]; }
    const float inv = 1.0f / s;
    #pragma unroll
    for (int l = 0; l < 4; ++l) lw[l] = e[l] * inv;
  }

  const float* fsrc[4] = {f0, f1, f2, f3};
  const float* fbase[4];
  int   cst[4], o00[4], o10[4], o01[4], o11[4];
  float w00[4], w10[4], w01[4], w11[4];
  #pragma unroll
  for (int l = 0; l < 4; ++l) {
    const int   Wl = 64 >> l;                // 64,32,16,8
    const float s  = (float)(4 << l);        // 4,8,16,32
    const float px = x / s - 0.5f, py = y / s - 0.5f;
    const float fx0 = floorf(px), fy0 = floorf(py);
    const float fx = px - fx0, fy = py - fy0;
    const int x0 = (int)fx0, y0 = (int)fy0, x1 = x0 + 1, y1 = y0 + 1;
    const float vx0 = (x0 >= 0 && x0 < Wl) ? 1.f : 0.f;
    const float vx1 = (x1 >= 0 && x1 < Wl) ? 1.f : 0.f;
    const float vy0 = (y0 >= 0 && y0 < Wl) ? 1.f : 0.f;
    const float vy1 = (y1 >= 0 && y1 < Wl) ? 1.f : 0.f;
    const int cx0 = min(max(x0, 0), Wl - 1), cx1 = min(max(x1, 0), Wl - 1);
    const int cy0 = min(max(y0, 0), Wl - 1), cy1 = min(max(y1, 0), Wl - 1);
    w00[l] = lw[l] * (1.f - fx) * (1.f - fy) * vx0 * vy0;
    w10[l] = lw[l] * fx * (1.f - fy) * vx1 * vy0;
    w01[l] = lw[l] * (1.f - fx) * fy * vx0 * vy1;
    w11[l] = lw[l] * fx * fy * vx1 * vy1;
    o00[l] = cy0 * Wl + cx0; o10[l] = cy0 * Wl + cx1;
    o01[l] = cy1 * Wl + cx0; o11[l] = cy1 * Wl + cx1;
    const long hw = (long)Wl * Wl;
    fbase[l] = fsrc[l] + (((long)b * 256 + g * 64) * 8 + t) * hw;
    cst[l]   = (int)(8 * hw);                // channel stride
  }

  for (int chunk = 0; chunk < 8; ++chunk) {
    #pragma unroll
    for (int j = 0; j < 8; ++j) {
      const int c = chunk * 8 + j;
      float acc = 0.0f;
      #pragma unroll
      for (int l = 0; l < 4; ++l) {
        const float* fb = fbase[l] + (long)c * cst[l];
        acc = fmaf(w00[l], fb[o00[l]], acc);
        acc = fmaf(w10[l], fb[o10[l]], acc);
        acc = fmaf(w01[l], fb[o01[l]], acc);
        acc = fmaf(w11[l], fb[o11[l]], acc);
      }
      sbuf[tid * 8 + j] = acc;
    }
    __syncthreads();
    // spatial: mean over t (8)  -> 32 sp x 8 ch = 256 threads
    {
      const int sp2 = tid >> 3, j2 = tid & 7;
      float s = 0.0f;
      #pragma unroll
      for (int tt = 0; tt < 8; ++tt) s += sbuf[(sp2 * 8 + tt) * 8 + j2];
      sf[(long)blk * 2048 + sp2 * 64 + chunk * 8 + j2] = s * 0.125f;
    }
    // temporal: mean over sp (32) -> 8 t x 8 ch = 64 threads
    if (tid < 64) {
      const int t2 = tid >> 3, j2 = tid & 7;
      float s = 0.0f;
      for (int ss = 0; ss < 32; ++ss) s += sbuf[(ss * 8 + t2) * 8 + j2];
      tfeat[(long)blk * 512 + t2 * 64 + chunk * 8 + j2] = s * 0.03125f;
    }
    __syncthreads();
  }
}

// ---------------------------------------------------------------------------
// Adaptive mixing per (b,q,g): out1 = relu(LN2(x @ M)); out2 = relu(LN2(S@out1))
// x: (P,64), M: (64,64), S: (O,P). Output written as (bq, (g*O+o)*64+d).
// ---------------------------------------------------------------------------
template<int P, int O, int PROW, int GOFF>
__global__ __launch_bounds__(256) void mix_kernel(
    const float* __restrict__ X, const float* __restrict__ PRM,
    float* __restrict__ OUT)
{
  __shared__ float sx[P * 64];
  __shared__ float sm[64 * 64];
  __shared__ float so1[P * 64];
  __shared__ float red[256];
  __shared__ float red2[256];

  const int blk = blockIdx.x;
  const int g   = blk & 3;
  const int bq  = blk >> 2;
  const int tid = threadIdx.x;

  const float* xg  = X + (long)blk * (P * 64);
  const float* prm = PRM + (long)bq * PROW + (long)g * GOFF;

  for (int i = tid; i < P * 64; i += 256) sx[i] = xg[i];
  for (int i = tid; i < 4096;   i += 256) sm[i] = prm[i];
  __syncthreads();

  // ---- out1[p][d] = sum_c x[p][c] * M[c][d]
  constexpr int E1 = (P * 64) / 256;
  float o1[E1];
  float lsum = 0.f, lsq = 0.f;
  #pragma unroll
  for (int j = 0; j < E1; ++j) {
    const int e = tid * E1 + j, p = e >> 6, d = e & 63;
    float acc = 0.f;
    for (int c = 0; c < 64; ++c) acc = fmaf(sx[p * 64 + c], sm[c * 64 + d], acc);
    o1[j] = acc; lsum += acc; lsq += acc * acc;
  }
  red[tid] = lsum; red2[tid] = lsq; __syncthreads();
  for (int s = 128; s > 0; s >>= 1) {
    if (tid < s) { red[tid] += red[tid + s]; red2[tid] += red2[tid + s]; }
    __syncthreads();
  }
  {
    const float inv_n = 1.0f / (float)(P * 64);
    const float mu  = red[0] * inv_n;
    const float var = red2[0] * inv_n - mu * mu;
    const float inv = rsqrtf(var + 1e-5f);
    __syncthreads();
    #pragma unroll
    for (int j = 0; j < E1; ++j) {
      const float v = (o1[j] - mu) * inv;
      so1[tid * E1 + j] = v > 0.f ? v : 0.f;
    }
  }
  __syncthreads();

  // ---- out2[o][d] = sum_p S[o][p] * out1[p][d]  (each thread owns one o)
  constexpr int E2 = (O * 64) / 256;
  const int o  = (tid * E2) >> 6;
  const int d0 = (tid * E2) & 63;
  const float* Srow = prm + 4096 + o * P;
  float sreg[P];
  #pragma unroll
  for (int p = 0; p < P; ++p) sreg[p] = Srow[p];

  float o2[E2];
  lsum = 0.f; lsq = 0.f;
  #pragma unroll
  for (int j = 0; j < E2; ++j) {
    const int d = d0 + j;
    float acc = 0.f;
    #pragma unroll
    for (int p = 0; p < P; ++p) acc = fmaf(sreg[p], so1[p * 64 + d], acc);
    o2[j] = acc; lsum += acc; lsq += acc * acc;
  }
  red[tid] = lsum; red2[tid] = lsq; __syncthreads();
  for (int s = 128; s > 0; s >>= 1) {
    if (tid < s) { red[tid] += red[tid + s]; red2[tid] += red2[tid + s]; }
    __syncthreads();
  }
  const float inv_n = 1.0f / (float)(O * 64);
  const float mu  = red[0] * inv_n;
  const float var = red2[0] * inv_n - mu * mu;
  const float inv = rsqrtf(var + 1e-5f);
  float* outg = OUT + (long)bq * (4 * O * 64) + (long)(g * O) * 64;
  #pragma unroll
  for (int j = 0; j < E2; ++j) {
    const float v = (o2[j] - mu) * inv;
    outg[o * 64 + d0 + j] = v > 0.f ? v : 0.f;
  }
}

// ---------------------------------------------------------------------------
// Final: sum split-K partials + bias + residual, LayerNorm over D=256.
// One block per row.
// ---------------------------------------------------------------------------
__global__ __launch_bounds__(256) void resln_kernel(
    const float* __restrict__ Yp, int nchunks, long chunkStride,
    const float* __restrict__ bias, const float* __restrict__ Qr,
    const float* __restrict__ gam, const float* __restrict__ bet,
    float* __restrict__ OUT)
{
  __shared__ float red[256];
  __shared__ float red2[256];
  const int row = blockIdx.x;
  const int tid = threadIdx.x;
  float v = bias[tid] + Qr[(long)row * 256 + tid];
  for (int c = 0; c < nchunks; ++c) v += Yp[(long)c * chunkStride + (long)row * 256 + tid];
  red[tid] = v; red2[tid] = v * v; __syncthreads();
  for (int s = 128; s > 0; s >>= 1) {
    if (tid < s) { red[tid] += red[tid + s]; red2[tid] += red2[tid + s]; }
    __syncthreads();
  }
  const float mu  = red[0] * (1.0f / 256.0f);
  const float var = red2[0] * (1.0f / 256.0f) - mu * mu;
  OUT[(long)row * 256 + tid] = (v - mu) * rsqrtf(var + 1e-5f) * gam[tid] + bet[tid];
}

// ---------------------------------------------------------------------------
extern "C" void kernel_launch(void* const* d_in, const int* in_sizes, int n_in,
                              void* d_out, int out_size, void* d_ws, size_t ws_size,
                              hipStream_t stream)
{
  (void)in_sizes; (void)n_in; (void)out_size; (void)ws_size;
  const float* feat0 = (const float*)d_in[0];
  const float* feat1 = (const float*)d_in[1];
  const float* feat2 = (const float*)d_in[2];
  const float* feat3 = (const float*)d_in[3];
  const float* boxes = (const float*)d_in[4];
  const float* sq    = (const float*)d_in[5];
  const float* tq    = (const float*)d_in[6];
  const float* W_off = (const float*)d_in[7];
  const float* b_off = (const float*)d_in[8];
  const float* W_ps  = (const float*)d_in[9];
  const float* b_ps  = (const float*)d_in[10];
  const float* Wo_s  = (const float*)d_in[11];
  const float* bo_s  = (const float*)d_in[12];
  const float* g_s   = (const float*)d_in[13];
  const float* be_s  = (const float*)d_in[14];
  const float* W_pt  = (const float*)d_in[15];
  const float* b_pt  = (const float*)d_in[16];
  const float* Wo_t  = (const float*)d_in[17];
  const float* bo_t  = (const float*)d_in[18];
  const float* g_t   = (const float*)d_in[19];
  const float* be_t  = (const float*)d_in[20];
  float* out = (float*)d_out;
  float* ws  = (float*)d_ws;

  float* off = ws;                       // 400*384
  float* sf  = off + 153600;             // 400*4*32*64
  float* tfb = sf  + 3276800;            // 400*4*8*64
  float* ps  = tfb + 819200;             // 400*32768
  float* pt  = ps  + 13107200;           // 400*17408
  float* ms  = pt  + 6963200;            // 400*32768
  float* mtb = ms  + 13107200;           // 400*8192
  float* gs  = mtb + 3276800;            // 16 * 400*256 split-K partials
  float* gt  = gs  + 1638400;            // 16 * 400*256

  auto gemm = [&](const float* A, const float* Bm, const float* bias, float* C,
                  int M, int N, int K, int kc) {
    const int tiles  = (M / 16) * (N / 64);
    const int waves  = tiles * kc;
    const int blocks = (waves + GEMM_WAVES - 1) / GEMM_WAVES;
    gemm_wmma_kernel<<<blocks, 32 * GEMM_WAVES, 0, stream>>>(A, Bm, bias, C, M, N, K, kc);
  };

  // 1) sampling offsets
  gemm(sq, W_off, b_off, off, 400, 384, 256, 1);
  // 2) multi-level bilinear sampling -> spatial/temporal features
  sample_kernel<<<1600, 256, 0, stream>>>(feat0, feat1, feat2, feat3, boxes, off, sf, tfb);
  // 3) per-query mixing parameters (the big weight-streaming GEMMs)
  gemm(sq, W_ps, b_ps, ps, 400, 32768, 256, 1);
  gemm(tq, W_pt, b_pt, pt, 400, 17408, 256, 1);
  // 4) adaptive mixing (per-group small GEMMs + LN2 + ReLU)
  mix_kernel<32, 128, 32768, 8192><<<1600, 256, 0, stream>>>(sf,  ps, ms);
  mix_kernel< 8,  32, 17408, 4352><<<1600, 256, 0, stream>>>(tfb, pt, mtb);
  // 5) output projections, split-K=16 for wave parallelism (100 tiles -> 1600 waves)
  gemm(ms,  Wo_s, nullptr, gs, 400, 256, 32768, 16);
  gemm(mtb, Wo_t, nullptr, gt, 400, 256, 8192, 16);
  // 6) partial-sum + bias + residual + LayerNorm
  resln_kernel<<<400, 256, 0, stream>>>(gs, 16, 102400L, bo_s, sq, g_s, be_s, out);
  resln_kernel<<<400, 256, 0, stream>>>(gt, 16, 102400L, bo_t, tq, g_t, be_t, out + 102400);
}
